// HistoryEncoder_35519379538076
// MI455X (gfx1250) — matrix operand order repaired and checked by reading, not compiled, not run on hardware
//
#include <hip/hip_runtime.h>
#include <hip/hip_bf16.h>
#include <math.h>

// ---------------------------------------------------------------------------
// Problem constants (reference): B=4, M=2048, IN_DIM=512, D=1024, H=4, NL=2
// ---------------------------------------------------------------------------
#define D_MODEL 1024
#define IN_DIM  512
#define SEQ     2048
#define NBATCH  4
#define NHEAD   4
#define HEADDIM 256
#define HDHALF  128
#define NLAYER  2
#define ROWS    (NBATCH * SEQ)   // 8192 token rows

typedef __attribute__((ext_vector_type(16))) __bf16 v16bf;
typedef __attribute__((ext_vector_type(8)))  __bf16 v8bf;
typedef __attribute__((ext_vector_type(4)))  __bf16 v4bf;
typedef __attribute__((ext_vector_type(8)))  float  v8f;
typedef __attribute__((__vector_size__(4 * sizeof(int)))) int vi4;

__device__ __forceinline__ v8f wmma_bf16(v16bf a, v16bf b, v8f c) {
  // 8 args: (neg_a, A, neg_b, B, c_mod, C, reuse_a, reuse_b)
  return __builtin_amdgcn_wmma_f32_16x16x32_bf16(false, a, false, b, (short)0, c, false, false);
}

__device__ __forceinline__ v16bf make_v16(v8bf lo, v8bf hi) {
  union { v16bf v; v8bf h[2]; } u;
  u.h[0] = lo; u.h[1] = hi;
  return u.v;
}

// A-matrix 16x32 bf16 fragment (ISA 7.12.2): lane L holds row (L&15).
// Lane half 0: k = [0..15]; lane half 1: k = [16..31] (contiguous, matches B).
__device__ __forceinline__ v16bf load_a_frag(const __bf16* rowptr, int kbase, int lhalf) {
  const int h16 = lhalf * 16;
  v8bf lo = *(const v8bf*)(rowptr + kbase + h16);
  v8bf hi = *(const v8bf*)(rowptr + kbase + h16 + 8);
  return make_v16(lo, hi);
}

// ---------------------------------------------------------------------------
// Async global->LDS 16B copy (ASYNCcnt path); fallback = vector copy.
// Builtin parameter type is a generic int4* (per hipcc diagnostic).
// ---------------------------------------------------------------------------
#if defined(__has_builtin)
#if __has_builtin(__builtin_amdgcn_global_load_async_to_lds_b128)
#define HAVE_ASYNC_LDS 1
#endif
#if __has_builtin(__builtin_amdgcn_s_wait_asynccnt)
#define HAVE_WAIT_ASYNC 1
#endif
#endif

__device__ __forceinline__ void async_copy16(const __bf16* g, __bf16* l) {
#ifdef HAVE_ASYNC_LDS
  __builtin_amdgcn_global_load_async_to_lds_b128((vi4*)(void*)g, (vi4*)(void*)l, 0, 0);
#else
  *(v8bf*)l = *(const v8bf*)g;
#endif
}

__device__ __forceinline__ void wait_async_lds() {
#ifdef HAVE_ASYNC_LDS
#ifdef HAVE_WAIT_ASYNC
  __builtin_amdgcn_s_wait_asynccnt(0);
#else
  asm volatile("s_wait_asynccnt 0" ::: "memory");
#endif
#endif
}

// ---------------------------------------------------------------------------
// GEMM: out = act(A[M,K](bf16) @ Bw[K,N](bf16) + bias) [+ residual]
// fp32 accumulate via WMMA. Block tile 128x128x32, 256 threads = 8 waves,
// each wave computes a 64x32 sub-tile (4x2 WMMA tiles).
// A tile staged via async global->LDS; B tile transposed [n][k] via ds stores.
// ---------------------------------------------------------------------------
#define BM 128
#define BN 128
#define BK 32
#define LDA_S 40
#define LDB_S 40

__global__ __launch_bounds__(256) void gemm_bf16_wmma(
    const __bf16* __restrict__ A, const __bf16* __restrict__ Bw,
    float* __restrict__ Cf, __bf16* __restrict__ Cb,
    const float* __restrict__ bias, const float* __restrict__ residual,
    int M, int N, int K, int gelu_act)
{
  __shared__ __align__(16) __bf16 sA[BM * LDA_S];
  __shared__ __align__(16) __bf16 sB[BN * LDB_S];

  const int tid   = threadIdx.x;
  const int lane  = tid & 31;
  const int l15   = lane & 15;
  const int lhalf = lane >> 4;
  const int wave  = tid >> 5;
  const int wr    = wave >> 2;   // 0..1
  const int wc    = wave & 3;    // 0..3
  const int m0    = blockIdx.y * BM;
  const int n0    = blockIdx.x * BN;

  const v8f zero8 = {0.f, 0.f, 0.f, 0.f, 0.f, 0.f, 0.f, 0.f};
  v8f acc[4][2];
#pragma unroll
  for (int i = 0; i < 4; ++i)
#pragma unroll
    for (int j = 0; j < 2; ++j) acc[i][j] = zero8;

  for (int kt = 0; kt < K; kt += BK) {
#pragma unroll
    for (int i = 0; i < 2; ++i) {
      const int c = tid + i * 256;         // 512 x 16B chunks per tile
      // A tile: row-major, async straight into LDS
      const int ar = c >> 2, ac = (c & 3) << 3;
      async_copy16(A + (size_t)(m0 + ar) * K + kt + ac, sA + ar * LDA_S + ac);
      // B tile: load 8 bf16 along n, scatter-transpose into [n][k]
      const int kr = c >> 4, nb = (c & 15) << 3;
      const v8bf b8 = *(const v8bf*)(Bw + (size_t)(kt + kr) * N + n0 + nb);
#pragma unroll
      for (int j = 0; j < 8; ++j) sB[(nb + j) * LDB_S + kr] = b8[j];
      if (kt + BK < K) // next K-tile -> global_prefetch_b8
        __builtin_prefetch(Bw + (size_t)(kt + BK + kr) * N + n0 + nb, 0, 1);
    }
    wait_async_lds();
    __syncthreads();

    v16bf afr[4], bfr[2];
#pragma unroll
    for (int sm = 0; sm < 4; ++sm)
      afr[sm] = load_a_frag(sA + (wr * 64 + sm * 16 + l15) * LDA_S, 0, lhalf);
#pragma unroll
    for (int sn = 0; sn < 2; ++sn)
      bfr[sn] = *(const v16bf*)(sB + (wc * 32 + sn * 16 + l15) * LDB_S + lhalf * 16);
#pragma unroll
    for (int sm = 0; sm < 4; ++sm)
#pragma unroll
      for (int sn = 0; sn < 2; ++sn)
        acc[sm][sn] = wmma_bf16(afr[sm], bfr[sn], acc[sm][sn]);
    __syncthreads();
  }

  // epilogue: C/D layout -> VGPR j holds row (lhalf*8 + j), column = l15
#pragma unroll
  for (int sm = 0; sm < 4; ++sm) {
#pragma unroll
    for (int sn = 0; sn < 2; ++sn) {
#pragma unroll
      for (int j = 0; j < 8; ++j) {
        const int row = m0 + wr * 64 + sm * 16 + lhalf * 8 + j;
        const int col = n0 + wc * 32 + sn * 16 + l15;
        float v = acc[sm][sn][j];
        if (bias) v += bias[col];
        if (gelu_act) v = 0.5f * v * (1.0f + erff(v * 0.70710678118654752f));
        const size_t idx = (size_t)row * N + col;
        if (Cf) {
          float o = v;
          if (residual) o += residual[idx];
          Cf[idx] = o;
        }
        if (Cb) Cb[idx] = (__bf16)v;
      }
    }
  }
}

// ---------------------------------------------------------------------------
// Flash attention over bf16 qkv: block = 4 waves, 64 q rows, one (batch,head).
// Q/K staged via async global->LDS (already bf16, layout-compatible);
// V staged transposed [hd][key]; P restaged per-wave with s_wait_dscnt.
// ---------------------------------------------------------------------------
#define QS 264
#define KS 264
#define VS 40
#define PS 40

__global__ __launch_bounds__(128) void flash_attn_k(
    const __bf16* __restrict__ qkv, const float* __restrict__ mask, __bf16* __restrict__ ctx)
{
  __shared__ __align__(16) __bf16 sQ[64 * QS];
  __shared__ __align__(16) __bf16 sK[32 * KS];
  __shared__ __align__(16) __bf16 sV[HEADDIM * VS];
  __shared__ __align__(16) __bf16 sP[4 * 16 * PS];

  const int tid   = threadIdx.x;
  const int lane  = tid & 31;
  const int l15   = lane & 15;
  const int lhalf = lane >> 4;
  const int wave  = tid >> 5;
  const int qb    = blockIdx.x;
  const int hh    = blockIdx.y;
  const int bb    = blockIdx.z;

  const size_t rstride = 3 * D_MODEL;
  const size_t rowbase = (size_t)bb * SEQ;

  // stage Q tile (64 x 256 bf16) -- async, no transform needed
#pragma unroll
  for (int i = 0; i < 16; ++i) {
    const int c = tid + i * 128;             // 2048 x 8-elem chunks
    const int r = c >> 5;
    const int col = (c & 31) << 3;
    async_copy16(qkv + (rowbase + qb * 64 + r) * rstride + hh * HEADDIM + col,
                 sQ + r * QS + col);
  }

  const v8f zero8 = {0.f, 0.f, 0.f, 0.f, 0.f, 0.f, 0.f, 0.f};
  v8f O[16];
#pragma unroll
  for (int s = 0; s < 16; ++s) O[s] = zero8;
  float mrow[8], lrow[8];
#pragma unroll
  for (int j = 0; j < 8; ++j) { mrow[j] = -1e30f; lrow[j] = 0.f; }

  for (int kb = 0; kb < SEQ / 32; ++kb) {
    __syncthreads();   // previous iteration done reading sK/sV
    const int k0 = kb * 32;

    // stage K (32x256 row-major, async) + V (transposed [hd][key])
#pragma unroll
    for (int i = 0; i < 8; ++i) {
      const int c = tid + i * 128;           // 1024 x 8-elem chunks
      const int r = c >> 5;
      const int col = (c & 31) << 3;
      const __bf16* kp = qkv + (rowbase + k0 + r) * rstride + D_MODEL + hh * HEADDIM + col;
      async_copy16(kp, sK + r * KS + col);
      const v8bf v8 = *(const v8bf*)(kp + D_MODEL);
#pragma unroll
      for (int j = 0; j < 8; ++j) sV[(col + j) * VS + r] = v8[j];
    }
    wait_async_lds();
    __syncthreads();

    // S = Q @ K^T (16 q-rows x 32 keys), K-loop over 8 hd chunks
    v8f S0 = zero8, S1 = zero8;
    const __bf16* qrow = sQ + (wave * 16 + l15) * QS;
#pragma unroll
    for (int hc = 0; hc < 8; ++hc) {
      v16bf a  = load_a_frag(qrow, hc * 32, lhalf);
      v16bf b0 = *(const v16bf*)(sK + l15 * KS + hc * 32 + lhalf * 16);
      v16bf b1 = *(const v16bf*)(sK + (16 + l15) * KS + hc * 32 + lhalf * 16);
      S0 = wmma_bf16(a, b0, S0);
      S1 = wmma_bf16(a, b1, S1);
    }

    const float mk0 = mask[rowbase + k0 + l15];
    const float mk1 = mask[rowbase + k0 + 16 + l15];

    __bf16* pw = sP + wave * 16 * PS;
#pragma unroll
    for (int j = 0; j < 8; ++j) {
      float s0 = S0[j] * 0.0625f;            // 1/sqrt(256)
      float s1 = S1[j] * 0.0625f;
      if (mk0 == 0.f) s0 = -1e30f;
      if (mk1 == 0.f) s1 = -1e30f;
      float rm = fmaxf(s0, s1);              // 16-lane row reductions (wave32)
      rm = fmaxf(rm, __shfl_xor(rm, 1, 32));
      rm = fmaxf(rm, __shfl_xor(rm, 2, 32));
      rm = fmaxf(rm, __shfl_xor(rm, 4, 32));
      rm = fmaxf(rm, __shfl_xor(rm, 8, 32));
      const float nm   = fmaxf(mrow[j], rm);
      const float corr = __expf(mrow[j] - nm);
      const float p0 = __expf(s0 - nm);
      const float p1 = __expf(s1 - nm);
      float rs = p0 + p1;
      rs += __shfl_xor(rs, 1, 32);
      rs += __shfl_xor(rs, 2, 32);
      rs += __shfl_xor(rs, 4, 32);
      rs += __shfl_xor(rs, 8, 32);
      lrow[j] = lrow[j] * corr + rs;
      mrow[j] = nm;
#pragma unroll
      for (int s = 0; s < 16; ++s) O[s][j] *= corr;
      pw[(lhalf * 8 + j) * PS + l15]      = (__bf16)p0;
      pw[(lhalf * 8 + j) * PS + 16 + l15] = (__bf16)p1;
    }
    asm volatile("s_wait_dscnt 0" ::: "memory");  // own-wave P stores visible

    // O += P @ V
    v16bf ap = load_a_frag(pw + l15 * PS, 0, lhalf);
#pragma unroll
    for (int s = 0; s < 16; ++s) {
      v16bf bv = *(const v16bf*)(sV + (s * 16 + l15) * VS + lhalf * 16);
      O[s] = wmma_bf16(ap, bv, O[s]);
    }
  }

  // normalize and write context (bf16, feeds o-proj GEMM)
#pragma unroll
  for (int j = 0; j < 8; ++j) {
    const float inv = (lrow[j] > 0.f) ? (1.0f / lrow[j]) : 0.f;
    const int qr = qb * 64 + wave * 16 + lhalf * 8 + j;
    __bf16* op = ctx + (rowbase + qr) * D_MODEL + hh * HEADDIM;
#pragma unroll
    for (int s = 0; s < 16; ++s) op[s * 16 + l15] = (__bf16)(O[s][j] * inv);
  }
}

// ---------------------------------------------------------------------------
// LayerNorm: fp32 in (residual stream), bf16 out (feeds GEMM A)
// ---------------------------------------------------------------------------
__global__ __launch_bounds__(256) void layernorm_k(
    const float* __restrict__ x, const float* __restrict__ g, const float* __restrict__ bt,
    __bf16* __restrict__ out)
{
  const int row = blockIdx.x;
  const int tid = threadIdx.x;
  const float4 v = *(const float4*)(x + (size_t)row * D_MODEL + tid * 4);
  float s  = v.x + v.y + v.z + v.w;
  float s2 = v.x * v.x + v.y * v.y + v.z * v.z + v.w * v.w;
#pragma unroll
  for (int m = 1; m < 32; m <<= 1) { s += __shfl_xor(s, m, 32); s2 += __shfl_xor(s2, m, 32); }
  __shared__ float ws1[8], ws2[8];
  if ((tid & 31) == 0) { ws1[tid >> 5] = s; ws2[tid >> 5] = s2; }
  __syncthreads();
  s = 0.f; s2 = 0.f;
#pragma unroll
  for (int i = 0; i < 8; ++i) { s += ws1[i]; s2 += ws2[i]; }
  const float mu  = s * (1.0f / D_MODEL);
  const float var = s2 * (1.0f / D_MODEL) - mu * mu;
  const float r   = rsqrtf(var + 1e-5f);
  const float4 gv = *(const float4*)(g + tid * 4);
  const float4 bv = *(const float4*)(bt + tid * 4);
  v4bf o = { (__bf16)((v.x - mu) * r * gv.x + bv.x),
             (__bf16)((v.y - mu) * r * gv.y + bv.y),
             (__bf16)((v.z - mu) * r * gv.z + bv.z),
             (__bf16)((v.w - mu) * r * gv.w + bv.w) };
  *(v4bf*)(out + (size_t)row * D_MODEL + tid * 4) = o;
}

// ---------------------------------------------------------------------------
// RoPE in-place on bf16 q,k slices of packed qkv (fp32 math)
// ---------------------------------------------------------------------------
__global__ __launch_bounds__(256) void rope_k(__bf16* __restrict__ qkv, const int* __restrict__ pos)
{
  const int idx  = blockIdx.x * 256 + threadIdx.x;   // < B*M*H*HD2
  const int i    = idx & (HDHALF - 1);
  const int head = (idx >> 7) & (NHEAD - 1);
  const int bm   = idx >> 9;
  const float p   = (float)pos[bm];
  const float inv = __expf(-9.210340371976184f * ((float)(2 * i) * (1.0f / HEADDIM)));
  float sn, cs;
  __sincosf(p * inv, &sn, &cs);
  __bf16* q = qkv + (size_t)bm * (3 * D_MODEL) + head * HEADDIM;
  float x1 = (float)q[i], x2 = (float)q[i + HDHALF];
  q[i]          = (__bf16)(x1 * cs - x2 * sn);
  q[i + HDHALF] = (__bf16)(x1 * sn + x2 * cs);
  __bf16* k = q + D_MODEL;
  x1 = (float)k[i]; x2 = (float)k[i + HDHALF];
  k[i]          = (__bf16)(x1 * cs - x2 * sn);
  k[i + HDHALF] = (__bf16)(x1 * sn + x2 * cs);
}

// ---------------------------------------------------------------------------
// fp32 -> bf16 cast (packed), one thread per float4 group
// ---------------------------------------------------------------------------
__global__ __launch_bounds__(256) void cast_f32_bf16_k(
    const float* __restrict__ in, __bf16* __restrict__ out, int nquads)
{
  const int i = blockIdx.x * 256 + threadIdx.x;
  if (i < nquads) {
    const float4 f = *(const float4*)(in + (size_t)i * 4);
    v4bf o = { (__bf16)f.x, (__bf16)f.y, (__bf16)f.z, (__bf16)f.w };
    *(v4bf*)(out + (size_t)i * 4) = o;
  }
}

// ---------------------------------------------------------------------------
// Final gather: lengths = sum(mask); out[b] = len>0 ? h[b, len-1] : start
// ---------------------------------------------------------------------------
__global__ __launch_bounds__(256) void gather_k(
    const float* __restrict__ h, const float* __restrict__ mask,
    const float* __restrict__ start, float* __restrict__ out)
{
  const int b = blockIdx.x;
  const int tid = threadIdx.x;
  float s = 0.f;
  for (int i = tid; i < SEQ; i += 256) s += mask[(size_t)b * SEQ + i];
#pragma unroll
  for (int m = 1; m < 32; m <<= 1) s += __shfl_xor(s, m, 32);
  __shared__ float ws[8];
  if ((tid & 31) == 0) ws[tid >> 5] = s;
  __syncthreads();
  float tot = 0.f;
#pragma unroll
  for (int i = 0; i < 8; ++i) tot += ws[i];
  const int len = (int)(tot + 0.5f);
  for (int d = tid; d < D_MODEL; d += 256)
    out[(size_t)b * D_MODEL + d] =
        (len > 0) ? h[((size_t)b * SEQ + (len - 1)) * D_MODEL + d] : start[d];
}

// ---------------------------------------------------------------------------
extern "C" void kernel_launch(void* const* d_in, const int* in_sizes, int n_in,
                              void* d_out, int out_size, void* d_ws, size_t ws_size,
                              hipStream_t stream) {
  const float* x          = (const float*)d_in[0];
  const int*   positions  = (const int*)  d_in[1];
  const float* mask       = (const float*)d_in[2];
  const float* in_w       = (const float*)d_in[3];
  const float* in_b       = (const float*)d_in[4];
  const float* qkv_w      = (const float*)d_in[5];
  const float* o_w        = (const float*)d_in[6];
  const float* ln1_g      = (const float*)d_in[7];
  const float* ln1_b      = (const float*)d_in[8];
  const float* ffn_w1     = (const float*)d_in[9];
  const float* ffn_b1     = (const float*)d_in[10];
  const float* ffn_w2     = (const float*)d_in[11];
  const float* ffn_b2     = (const float*)d_in[12];
  const float* ln2_g      = (const float*)d_in[13];
  const float* ln2_b      = (const float*)d_in[14];
  const float* start      = (const float*)d_in[15];
  float* out = (float*)d_out;

  // ---- workspace layout: fp32 residual stream + bf16 everything else ----
  char* w = (char*)d_ws;
  size_t off = 0;
  float* h = (float*)(w + off);        off += (size_t)ROWS * D_MODEL * 4;       // 32 MB
  __bf16* qkvb = (__bf16*)(w + off);   off += (size_t)ROWS * 3 * D_MODEL * 2;   // 48 MB
  __bf16* hnb  = (__bf16*)(w + off);   off += (size_t)ROWS * D_MODEL * 2;       // 16 MB
  __bf16* ctxb = (__bf16*)(w + off);   off += (size_t)ROWS * D_MODEL * 2;       // 16 MB
  __bf16* f1b  = (__bf16*)(w + off);   off += (size_t)ROWS * 4 * D_MODEL * 2;   // 64 MB
  __bf16* xb   = (__bf16*)(w + off);   off += (size_t)ROWS * IN_DIM * 2;        // 8 MB
  __bf16* w_in = (__bf16*)(w + off);   off += (size_t)IN_DIM * D_MODEL * 2;
  __bf16* w_qkv= (__bf16*)(w + off);   off += (size_t)NLAYER * D_MODEL * 3 * D_MODEL * 2;
  __bf16* w_o  = (__bf16*)(w + off);   off += (size_t)NLAYER * D_MODEL * D_MODEL * 2;
  __bf16* w_f1 = (__bf16*)(w + off);   off += (size_t)NLAYER * D_MODEL * 4 * D_MODEL * 2;
  __bf16* w_f2 = (__bf16*)(w + off);   off += (size_t)NLAYER * 4 * D_MODEL * D_MODEL * 2;

  // ---- one-time fp32 -> bf16 casts (weights + input) ----
  auto cast = [&](const float* src, __bf16* dst, size_t nelem) {
    const int nq = (int)(nelem / 4);
    cast_f32_bf16_k<<<(nq + 255) / 256, 256, 0, stream>>>(src, dst, nq);
  };
  cast(x,      xb,    (size_t)ROWS * IN_DIM);
  cast(in_w,   w_in,  (size_t)IN_DIM * D_MODEL);
  cast(qkv_w,  w_qkv, (size_t)NLAYER * D_MODEL * 3 * D_MODEL);
  cast(o_w,    w_o,   (size_t)NLAYER * D_MODEL * D_MODEL);
  cast(ffn_w1, w_f1,  (size_t)NLAYER * D_MODEL * 4 * D_MODEL);
  cast(ffn_w2, w_f2,  (size_t)NLAYER * 4 * D_MODEL * D_MODEL);

  // h = x @ in_w + in_b                               [8192x512 @ 512x1024]
  gemm_bf16_wmma<<<dim3(D_MODEL / BN, ROWS / BM), 256, 0, stream>>>(
      xb, w_in, h, nullptr, in_b, nullptr, ROWS, D_MODEL, IN_DIM, 0);

  for (int l = 0; l < NLAYER; ++l) {
    // hn = LN1(h)  (bf16 out)
    layernorm_k<<<ROWS, 256, 0, stream>>>(h, ln1_g + l * D_MODEL, ln1_b + l * D_MODEL, hnb);
    // qkv = hn @ qkv_w[l]  (bf16 out)                 [8192x1024 @ 1024x3072]
    gemm_bf16_wmma<<<dim3(3 * D_MODEL / BN, ROWS / BM), 256, 0, stream>>>(
        hnb, w_qkv + (size_t)l * D_MODEL * 3 * D_MODEL, nullptr, qkvb,
        nullptr, nullptr, ROWS, 3 * D_MODEL, D_MODEL, 0);
    // RoPE in place on q,k
    rope_k<<<(NBATCH * SEQ * NHEAD * HDHALF) / 256, 256, 0, stream>>>(qkvb, positions);
    // ctx = softmax(QK^T/16) V   (flash, fused, bf16 out)
    flash_attn_k<<<dim3(SEQ / 64, NHEAD, NBATCH), 128, 0, stream>>>(qkvb, mask, ctxb);
    // h = h + ctx @ o_w[l]                            [8192x1024 @ 1024x1024]
    gemm_bf16_wmma<<<dim3(D_MODEL / BN, ROWS / BM), 256, 0, stream>>>(
        ctxb, w_o + (size_t)l * D_MODEL * D_MODEL, h, nullptr,
        nullptr, h, ROWS, D_MODEL, D_MODEL, 0);
    // hn = LN2(h)
    layernorm_k<<<ROWS, 256, 0, stream>>>(h, ln2_g + l * D_MODEL, ln2_b + l * D_MODEL, hnb);
    // f1 = gelu(hn @ ffn_w1[l] + b1)  (bf16 out)      [8192x1024 @ 1024x4096]
    gemm_bf16_wmma<<<dim3(4 * D_MODEL / BN, ROWS / BM), 256, 0, stream>>>(
        hnb, w_f1 + (size_t)l * D_MODEL * 4 * D_MODEL, nullptr, f1b,
        ffn_b1 + (size_t)l * 4 * D_MODEL, nullptr, ROWS, 4 * D_MODEL, D_MODEL, 1);
    // h = h + f1 @ ffn_w2[l] + b2                     [8192x4096 @ 4096x1024]
    gemm_bf16_wmma<<<dim3(D_MODEL / BN, ROWS / BM), 256, 0, stream>>>(
        f1b, w_f2 + (size_t)l * 4 * D_MODEL * D_MODEL, h, nullptr,
        ffn_b2 + (size_t)l * D_MODEL, h, ROWS, D_MODEL, 4 * D_MODEL, 0);
  }

  // final masked gather
  gather_k<<<NBATCH, 256, 0, stream>>>(h, mask, start, out);
}
